// SelfAttention_48687749268144
// MI455X (gfx1250) — matrix operand order repaired and checked by reading, not compiled, not run on hardware
//
#include <hip/hip_runtime.h>

typedef __attribute__((ext_vector_type(16))) __bf16 v16bf;
typedef __attribute__((ext_vector_type(8)))  __bf16 v8bf;
typedef __attribute__((ext_vector_type(8)))  float  v8f;

#define N_TOK 8192
#define NE    256
#define DH    64
#define WPB   4        // waves per block in attention kernel
#define BKEY  64       // keys processed per iteration
#define KP    72       // padded LDS row length (144 B rows: 16B-aligned, conflict-friendly)

static __device__ __forceinline__ v16bf make_frag(v8bf lo, v8bf hi) {
  union { v16bf v; v8bf h[2]; } u;
  u.h[0] = lo; u.h[1] = hi;
  return u.v;
}

static __device__ __forceinline__ v8f wmma_bf16(v16bf a, v16bf b, v8f c) {
  // D = A(16x32 bf16) * B(32x16 bf16) + C(16x16 f32)
  return __builtin_amdgcn_wmma_f32_16x16x32_bf16(false, a, false, b, (short)0, c,
                                                 false, false);
}

// ---- CDNA5 async global->LDS copy (ASYNCcnt path), 16 bytes per lane ----
static __device__ __forceinline__ void async_ld_b128(const void* lds_dst, const void* gsrc) {
  const unsigned loff = (unsigned)(uintptr_t)lds_dst;          // low 32 bits = LDS offset
  const unsigned long long ga = (unsigned long long)(uintptr_t)gsrc;
  asm volatile("global_load_async_to_lds_b128 %0, %1, off"
               :: "v"(loff), "v"(ga) : "memory");
}
static __device__ __forceinline__ void wait_async0() {
  asm volatile("s_wait_asynccnt 0" ::: "memory");
}

// A fragment (16x32): lane<16 -> row=row0+lane, k = k0 + {0..7,16..23}
//                     lane>=16 -> row=row0+lane-16, k = k0 + {8..15,24..31}
static __device__ __forceinline__ v16bf a_frag_from_f32(const float* __restrict__ base,
                                                        int ld, int row0, int k0, int lane) {
  const int r  = row0 + (lane & 15);
  const int kb = k0 + ((lane & 16) ? 8 : 0);
  const float* p = base + r * ld + kb;
  union { v16bf v; __bf16 e[16]; } u;
#pragma unroll
  for (int i = 0; i < 8; ++i) u.e[i]     = (__bf16)p[i];        // k = kb..kb+7
#pragma unroll
  for (int i = 0; i < 8; ++i) u.e[8 + i] = (__bf16)p[16 + i];   // k = kb+16..kb+23
  return u.v;
}

// B fragment (32x16): lane holds col n = n0+(lane&15), k = k0 + {0..7,16..23}
// (lane<16) or k0 + {8..15,24..31} (lane>=16).  B[k][n] = W[k][n], W f32 row-major, leading dim ld.
static __device__ __forceinline__ v16bf b_frag_from_f32(const float* __restrict__ base,
                                                        int ld, int k0, int n0, int lane) {
  const int n  = n0 + (lane & 15);
  const int kb = k0 + ((lane & 16) ? 8 : 0);
  union { v16bf v; __bf16 e[16]; } u;
#pragma unroll
  for (int i = 0; i < 16; ++i) u.e[i] = (__bf16)base[(kb + i + (i & 8)) * ld + n];
  return u.v;
}

// ---------------- Phase 1: Q/K/V projections ----------------
// grid (512, 3); block 128 (4 waves, one 16x16 output tile each).
__global__ __launch_bounds__(128) void proj_kernel(const float* __restrict__ x,
                                                   const float* __restrict__ Wq,
                                                   const float* __restrict__ Wk,
                                                   const float* __restrict__ Wv,
                                                   __bf16* __restrict__ Qbf,
                                                   __bf16* __restrict__ Kbf,
                                                   __bf16* __restrict__ Vt) {
  const int lane = threadIdx.x & 31;
  const int wave = threadIdx.x >> 5;
  const int row0 = blockIdx.x * 16;
  const int mat  = blockIdx.y;                       // 0=Q 1=K 2=V
  const float* W = (mat == 0) ? Wq : (mat == 1) ? Wk : Wv;
  const int n0   = wave * 16;

  v8f c = {};
#pragma unroll
  for (int kc = 0; kc < NE; kc += 32) {
    v16bf a = a_frag_from_f32(x, NE, row0, kc, lane);
    v16bf b = b_frag_from_f32(W, DH, kc, n0, lane);
    c = wmma_bf16(a, b, c);
  }

  // C layout: VGPR r -> row r + 8*(lane>=16), col = lane&15
  const int colg  = n0 + (lane & 15);
  const int rbase = row0 + ((lane & 16) ? 8 : 0);
  if (mat == 2) {
    // store V transposed: Vt[feature][token]
#pragma unroll
    for (int r = 0; r < 8; ++r)
      Vt[(size_t)colg * N_TOK + (rbase + r)] = (__bf16)c[r];
  } else {
    __bf16* out = (mat == 0) ? Qbf : Kbf;
    const float scale = (mat == 0) ? (1.0f / 64.0f) : 1.0f;  // fold score scale into Q
#pragma unroll
    for (int r = 0; r < 8; ++r)
      out[(size_t)(rbase + r) * DH + colg] = (__bf16)(c[r] * scale);
  }
}

// ---------------- Phase 2: flash attention ----------------
// grid 128; block 128 (4 waves x 16 query rows = 64 queries/block).
// Double-buffered K/V staged with async global->LDS copies.
__global__ __launch_bounds__(WPB * 32) void attn_kernel(const __bf16* __restrict__ Qbf,
                                                        const __bf16* __restrict__ Kbf,
                                                        const __bf16* __restrict__ Vt,
                                                        float* __restrict__ out) {
  __shared__ __bf16 Klds[2][BKEY][KP];     // [buf][key][feature]
  __shared__ __bf16 Vlds[2][DH][KP];       // [buf][feature][key]
  __shared__ __bf16 Plds[WPB][16][KP];     // per-wave P scratch (C-layout -> A-layout)

  const int tid  = threadIdx.x;
  const int lane = tid & 31;
  const int wave = tid >> 5;
  const int half = (lane & 16) ? 8 : 0;
  const int lc   = lane & 15;
  const int row0 = blockIdx.x * (WPB * 16) + wave * 16;

  // Q fragments (Q already scaled by 1/64): A operand, K-chunks 0..31 and 32..63
  v16bf qa0, qa1;
  {
    const int kb = (lane & 16) ? 8 : 0;
    const __bf16* p = Qbf + (size_t)(row0 + lc) * DH;
    qa0 = make_frag(*(const v8bf*)(p + kb),      *(const v8bf*)(p + kb + 16));
    qa1 = make_frag(*(const v8bf*)(p + 32 + kb), *(const v8bf*)(p + 32 + kb + 16));
  }

  v8f acc[4] = {v8f{}, v8f{}, v8f{}, v8f{}};
  float m[8], l[8];
#pragma unroll
  for (int r = 0; r < 8; ++r) { m[r] = -3.0e38f; l[r] = 0.0f; }

  // async-stage one 64-key K/V chunk into buffer `buf`
  auto stage = [&](int buf, int j0) {
    for (int idx = tid; idx < BKEY * (DH / 8); idx += WPB * 32) {
      const int key = idx >> 3, fo = (idx & 7) * 8;
      async_ld_b128(&Klds[buf][key][fo], &Kbf[(size_t)(j0 + key) * DH + fo]);
    }
    for (int idx = tid; idx < DH * (BKEY / 8); idx += WPB * 32) {
      const int f = idx >> 3, ko = (idx & 7) * 8;
      async_ld_b128(&Vlds[buf][f][ko], &Vt[(size_t)f * N_TOK + j0 + ko]);
    }
  };

  stage(0, 0);
  const int nIter = N_TOK / BKEY;
  for (int it = 0; it < nIter; ++it) {
    const int buf = it & 1;
    wait_async0();          // our own async transfers into `buf` are complete
    __syncthreads();        // everyone's transfers visible; nobody still reads buf^1
    if (it + 1 < nIter) stage(buf ^ 1, (it + 1) * BKEY);   // prefetch next chunk

    // S = Q * K^T : 4 key subtiles of 16; B[k=feature][n=key] read from Klds rows
    v8f s[4];
#pragma unroll
    for (int t = 0; t < 4; ++t) {
      const __bf16* krow = &Klds[buf][t * 16 + lc][0];
      const int kb = (lane & 16) ? 8 : 0;
      v16bf b0 = make_frag(*(const v8bf*)(krow + kb),      *(const v8bf*)(krow + kb + 16));
      v16bf b1 = make_frag(*(const v8bf*)(krow + 32 + kb), *(const v8bf*)(krow + 32 + kb + 16));
      v8f c = {};
      c = wmma_bf16(qa0, b0, c);
      c = wmma_bf16(qa1, b1, c);
      s[t] = c;
    }

    // online softmax: per-row (M) stats; row r+half lives across the 16 lanes of a half
#pragma unroll
    for (int r = 0; r < 8; ++r) {
      float t0 = fmaxf(fmaxf(s[0][r], s[1][r]), fmaxf(s[2][r], s[3][r]));
      t0 = fmaxf(t0, __shfl_xor(t0, 1, 16));
      t0 = fmaxf(t0, __shfl_xor(t0, 2, 16));
      t0 = fmaxf(t0, __shfl_xor(t0, 4, 16));
      t0 = fmaxf(t0, __shfl_xor(t0, 8, 16));
      const float mn   = fmaxf(m[r], t0);
      const float corr = __expf(m[r] - mn);
      m[r] = mn;
      s[0][r] = __expf(s[0][r] - mn);
      s[1][r] = __expf(s[1][r] - mn);
      s[2][r] = __expf(s[2][r] - mn);
      s[3][r] = __expf(s[3][r] - mn);
      float rs = (s[0][r] + s[1][r]) + (s[2][r] + s[3][r]);
      rs += __shfl_xor(rs, 1, 16);
      rs += __shfl_xor(rs, 2, 16);
      rs += __shfl_xor(rs, 4, 16);
      rs += __shfl_xor(rs, 8, 16);
      l[r] = l[r] * corr + rs;
#pragma unroll
      for (int n = 0; n < 4; ++n) acc[n][r] *= corr;
    }

    // C-layout -> A-layout via per-wave LDS bounce (cross-lane, so explicit ds wait)
#pragma unroll
    for (int t = 0; t < 4; ++t)
#pragma unroll
      for (int r = 0; r < 8; ++r)
        Plds[wave][r + half][t * 16 + lc] = (__bf16)s[t][r];

    asm volatile("s_wait_dscnt 0" ::: "memory");

    v16bf pa0, pa1;
    {
      const int kb = (lane & 16) ? 8 : 0;
      const __bf16* p = &Plds[wave][lc][0];
      pa0 = make_frag(*(const v8bf*)(p + kb),      *(const v8bf*)(p + kb + 16));
      pa1 = make_frag(*(const v8bf*)(p + 32 + kb), *(const v8bf*)(p + 32 + kb + 16));
    }

    // acc += P(16x64) * V(64x64); B[k=key][n=feature] read from Vlds[feature] rows
#pragma unroll
    for (int n = 0; n < 4; ++n) {
      const __bf16* vrow = &Vlds[buf][n * 16 + lc][0];
      const int kb = (lane & 16) ? 8 : 0;
      v16bf b0 = make_frag(*(const v8bf*)(vrow + kb),      *(const v8bf*)(vrow + kb + 16));
      v16bf b1 = make_frag(*(const v8bf*)(vrow + 32 + kb), *(const v8bf*)(vrow + 32 + kb + 16));
      acc[n] = wmma_bf16(pa0, b0, acc[n]);
      acc[n] = wmma_bf16(pa1, b1, acc[n]);
    }
  }

  // normalize and store
  float inv[8];
#pragma unroll
  for (int r = 0; r < 8; ++r) inv[r] = 1.0f / l[r];
#pragma unroll
  for (int n = 0; n < 4; ++n)
#pragma unroll
    for (int r = 0; r < 8; ++r)
      out[(size_t)(row0 + half + r) * DH + n * 16 + lc] = acc[n][r] * inv[r];
}

extern "C" void kernel_launch(void* const* d_in, const int* in_sizes, int n_in,
                              void* d_out, int out_size, void* d_ws, size_t ws_size,
                              hipStream_t stream) {
  (void)in_sizes; (void)n_in; (void)out_size; (void)ws_size;
  const float* x  = (const float*)d_in[0];
  const float* Wq = (const float*)d_in[1];
  const float* Wk = (const float*)d_in[2];
  const float* Wv = (const float*)d_in[3];

  __bf16* Qbf = (__bf16*)d_ws;                        // [8192][64] bf16 (pre-scaled 1/64)
  __bf16* Kbf = Qbf + (size_t)N_TOK * DH;             // [8192][64] bf16
  __bf16* Vt  = Kbf + (size_t)N_TOK * DH;             // [64][8192] bf16 (transposed)
  float* out  = (float*)d_out;

  dim3 g1(N_TOK / 16, 3), b1(128);
  hipLaunchKernelGGL(proj_kernel, g1, b1, 0, stream, x, Wq, Wk, Wv, Qbf, Kbf, Vt);

  dim3 g2(N_TOK / (WPB * 16)), b2(WPB * 32);
  hipLaunchKernelGGL(attn_kernel, g2, b2, 0, stream, Qbf, Kbf, Vt, out);
}
